// MovieNet_25331717111820
// MI455X (gfx1250) — matrix operand order repaired and checked by reading, not compiled, not run on hardware
//
#include <hip/hip_runtime.h>
#include <hip/hip_bf16.h>

#define BATCH 32
#define SEQ   512
#define HID   512
#define MTOT  (BATCH*SEQ)   /* 16384 */
#define OUTD  7

typedef __attribute__((ext_vector_type(16))) _Float16 v16h;
typedef __attribute__((ext_vector_type(8)))  _Float16 v8h;
typedef __attribute__((ext_vector_type(8)))  float    v8f;

__device__ __forceinline__ v8f wmma16(v16h a, v16h b, v8f c) {
  return __builtin_amdgcn_wmma_f32_16x16x32_f16(false, a, false, b, (short)0, c, false, false);
}

// ---- CDNA5 async global->LDS copy (ASYNCcnt-tracked) ----------------------
__device__ __forceinline__ unsigned lds_off32(const void* p) {
  // generic pointer to LDS: low 32 bits are the group-segment byte offset
  return (unsigned)(unsigned long long)p;
}
__device__ __forceinline__ void async_copy_b128(unsigned lds_off, const void* gptr) {
  asm volatile("global_load_async_to_lds_b128 %0, %1, off"
               :: "v"(lds_off), "v"((unsigned long long)gptr)
               : "memory");
}
__device__ __forceinline__ void wait_async0() {
  asm volatile("s_wait_asynccnt 0x0" ::: "memory");
}

// A operand (16x32 f16) built from 32 consecutive f32 (global or LDS row).
// Per ISA: lane<16 holds K=[kb..kb+7, kb+16..kb+23]; lanes 16-31 kb+=8.
__device__ __forceinline__ v16h load_a_f32(const float* __restrict__ rowp, int kb) {
  float4 x0 = *(const float4*)(rowp + kb);
  float4 x1 = *(const float4*)(rowp + kb + 4);
  float4 x2 = *(const float4*)(rowp + kb + 16);
  float4 x3 = *(const float4*)(rowp + kb + 20);
  v16h a;
  a[0]=(_Float16)x0.x;  a[1]=(_Float16)x0.y;  a[2]=(_Float16)x0.z;  a[3]=(_Float16)x0.w;
  a[4]=(_Float16)x1.x;  a[5]=(_Float16)x1.y;  a[6]=(_Float16)x1.z;  a[7]=(_Float16)x1.w;
  a[8]=(_Float16)x2.x;  a[9]=(_Float16)x2.y;  a[10]=(_Float16)x2.z; a[11]=(_Float16)x2.w;
  a[12]=(_Float16)x3.x; a[13]=(_Float16)x3.y; a[14]=(_Float16)x3.z; a[15]=(_Float16)x3.w;
  return a;
}

// A operand from an f16 row (LDS or global).
__device__ __forceinline__ v16h load_a_h(const _Float16* rowp, int kb) {
  v8h lo = *(const v8h*)(rowp + kb);
  v8h hi = *(const v8h*)(rowp + kb + 16);
  v16h a;
#pragma unroll
  for (int i = 0; i < 8; ++i) { a[i] = lo[i]; a[8 + i] = hi[i]; }
  return a;
}

// B operand (32x16 f16): lane holds K=[kb..kb+7, kb+16..kb+23], kb = k + 8*(lane>>4), from W row (lane&15).
__device__ __forceinline__ v16h load_b_h(const _Float16* wrow, int hk) {
  int kb = hk - 8 * ((hk >> 4) & 1);
  v8h lo = *(const v8h*)(wrow + kb);
  v8h hi = *(const v8h*)(wrow + kb + 16);
  v16h b;
#pragma unroll
  for (int i = 0; i < 8; ++i) { b[i] = lo[i]; b[8 + i] = hi[i]; }
  return b;
}

// branch-free transcendentals (v_exp_f32 based)
__device__ __forceinline__ float sigm(float x) { return 1.0f / (1.0f + __expf(-x)); }
__device__ __forceinline__ float tanh_fast(float x) {
  float xc = fminf(12.0f, fmaxf(-12.0f, x));
  float e = __expf(2.0f * xc);
  return (e - 1.0f) / (e + 1.0f);
}

// ---------------------------------------------------------------------------
__global__ void cvt_f32_f16(const float* __restrict__ src, _Float16* __restrict__ dst, int n) {
  int i = blockIdx.x * blockDim.x + threadIdx.x;
  if (i < n) dst[i] = (_Float16)src[i];
}

// ---------------------------------------------------------------------------
// Fused projection: o{1,2,3}[row] = sum_n tanh( (X @ W^T)[row,n] + bias[n] ) * w{1,2,u}[n]
// block = 256 (8 waves), 16 rows per block, wave owns 4 N-tiles. A tile is
// async-staged (f32) into a double-buffered LDS tile shared by all 8 waves.
// ---------------------------------------------------------------------------
__global__ __launch_bounds__(256)
void proj_kernel(const float* __restrict__ X, const _Float16* __restrict__ Wh,
                 const float* __restrict__ bias, const float* __restrict__ pw1,
                 const float* __restrict__ pw2, const float* __restrict__ pwu,
                 int K, float* __restrict__ o1, float* __restrict__ o2, float* __restrict__ o3) {
  __shared__ float s_a[2][16][32];
  int tid = threadIdx.x;
  int lane = tid & 31, wv = tid >> 5;
  int laneLo = lane & 15, hi = lane >> 4;

  // stage 16x32 f32 A tile for k-slice into s_a[buf]: 128 threads x 16B
  auto stage = [&](int k, int buf) {
    if (tid < 128) {
      int r = tid >> 3, c = (tid & 7) * 4;
      async_copy_b128(lds_off32(&s_a[buf][r][c]),
                      X + ((size_t)blockIdx.x * 16 + r) * (size_t)K + k + c);
    }
  };

  v8f acc[4];
#pragma unroll
  for (int t = 0; t < 4; ++t) { v8f z = {}; acc[t] = z; }

  stage(0, 0);
  wait_async0();
  __syncthreads();

  int buf = 0;
  for (int k = 0; k < K; k += 32) {
    if (k + 32 < K) stage(k + 32, buf ^ 1);
    v16h a = load_a_f32(&s_a[buf][laneLo][0], hi * 8);
    int hk = k + hi * 16;
    if (k + 64 < K) __builtin_prefetch(Wh + (size_t)((wv * 4) * 16 + laneLo) * K + k + 64, 0, 1);
#pragma unroll
    for (int t = 0; t < 4; ++t) {
      int n = (wv * 4 + t) * 16 + laneLo;
      acc[t] = wmma16(a, load_b_h(Wh + (size_t)n * K, hk), acc[t]);
    }
    wait_async0();
    __syncthreads();
    buf ^= 1;
  }

  float pr0[8], pr1[8], pr2[8];
#pragma unroll
  for (int v = 0; v < 8; ++v) { pr0[v] = 0.f; pr1[v] = 0.f; pr2[v] = 0.f; }

#pragma unroll
  for (int t = 0; t < 4; ++t) {
    int n = (wv * 4 + t) * 16 + laneLo;
    float bb = bias[n], a1 = pw1[n], a2 = pw2[n], a3 = pwu[n];
#pragma unroll
    for (int v = 0; v < 8; ++v) {
      float val = tanh_fast(acc[t][v] + bb);
      pr0[v] += val * a1; pr1[v] += val * a2; pr2[v] += val * a3;
    }
  }

  __shared__ float red[16][4];
  if (tid < 64) ((float*)red)[tid] = 0.f;
  __syncthreads();
#pragma unroll
  for (int v = 0; v < 8; ++v) {
    int m = v + hi * 8;
    atomicAdd(&red[m][0], pr0[v]);
    atomicAdd(&red[m][1], pr1[v]);
    atomicAdd(&red[m][2], pr2[v]);
  }
  __syncthreads();
  if (tid < 16) {
    size_t gr = (size_t)blockIdx.x * 16 + tid;
    o1[gr] = red[tid][0]; o2[gr] = red[tid][1]; o3[gr] = red[tid][2];
  }
}

// ---------------------------------------------------------------------------
// 21 softmaxes over T. grid = (21 jobs, 32 batches), 256 threads (2 t each)
// ---------------------------------------------------------------------------
__global__ __launch_bounds__(256)
void softmax_kernel(const float* __restrict__ S, const float* __restrict__ b_att,
                    const float* __restrict__ b_ut, const float* __restrict__ b_uf,
                    const float* __restrict__ b_ua,
                    float* __restrict__ att_raw, float* __restrict__ uni_raw) {
  const int pa[15] = {0,0,0,0,0,1,1,1,1,2,2,2,3,3,4};
  const int pb[15] = {2,1,3,4,5,2,3,4,5,3,4,5,4,5,5};
  int job = blockIdx.x, b = blockIdx.y, tid = threadIdx.x;
  const float* p1; const float* p2 = nullptr; float bias;
  if (job < 15) {
    p1 = S + ((size_t)(pa[job]*3 + 0)) * MTOT + (size_t)b * SEQ;
    p2 = S + ((size_t)(pb[job]*3 + 1)) * MTOT + (size_t)b * SEQ;
    bias = b_att[0];
  } else {
    int m = job - 15;
    p1 = S + ((size_t)(m*3 + 2)) * MTOT + (size_t)b * SEQ;
    bias = (m == 0 || m == 3) ? b_ut[0] : ((m == 1 || m == 4) ? b_uf[0] : b_ua[0]);
  }
  float v0 = p1[tid]       + (p2 ? p2[tid]       : 0.f) + bias;
  float v1 = p1[tid + 256] + (p2 ? p2[tid + 256] : 0.f) + bias;

  __shared__ float red[256];
  red[tid] = fmaxf(v0, v1);
  __syncthreads();
  for (int s = 128; s > 0; s >>= 1) { if (tid < s) red[tid] = fmaxf(red[tid], red[tid + s]); __syncthreads(); }
  float mx = red[0];
  __syncthreads();
  float e0 = __expf(v0 - mx), e1 = __expf(v1 - mx);
  red[tid] = e0 + e1;
  __syncthreads();
  for (int s = 128; s > 0; s >>= 1) { if (tid < s) red[tid] += red[tid + s]; __syncthreads(); }
  float inv = 1.0f / red[0];
  float* dst = (job < 15 ? att_raw + (size_t)job * MTOT : uni_raw + (size_t)(job - 15) * MTOT)
               + (size_t)b * SEQ;
  dst[tid] = e0 * inv;
  dst[tid + 256] = e1 * inv;
}

// Reproduce concat(axis=-1).reshape(B,T,L) interleave
__global__ void gather_kernel(const float* __restrict__ att_raw, const float* __restrict__ uni_raw,
                              float* __restrict__ attn, float* __restrict__ enc_in) {
  int bt = blockIdx.x * blockDim.x + threadIdx.x;
  if (bt >= MTOT) return;
  int b = bt >> 9, t = bt & 511;
#pragma unroll
  for (int i = 0; i < 15; ++i) {
    int j = t * 15 + i;
    attn[(size_t)bt * 15 + i] = att_raw[(size_t)(j >> 9) * MTOT + (size_t)b * SEQ + (j & 511)];
  }
#pragma unroll
  for (int g = 0; g < 6; ++g) {
    int j = t * 6 + g;
    enc_in[(size_t)bt * 6 + g] = uni_raw[(size_t)(j >> 9) * MTOT + (size_t)b * SEQ + (j & 511)];
  }
}

// ---------------------------------------------------------------------------
// Encoder LSTM: single persistent workgroup, 32 waves, wave w owns j in
// [16w,16w+16) and computes gates {i,f,g,o} at N = g*512 + j. h in LDS (f16).
// ---------------------------------------------------------------------------
__global__ __launch_bounds__(1024)
void enc_lstm_kernel(const float* __restrict__ enc_in, const _Float16* __restrict__ whh,
                     const float* __restrict__ wih, const float* __restrict__ bias,
                     float* __restrict__ enc_out) {
  __shared__ _Float16 lds_h[32][HID];
  __shared__ float s_ein[32][8];
  int tid = threadIdx.x;
  int lane = tid & 31, wv = tid >> 5;
  int laneLo = lane & 15, hi = lane >> 4;
  int j = wv * 16 + laneLo;

  float wih6[4][6], bg[4];
#pragma unroll
  for (int g = 0; g < 4; ++g) {
    int n = g * HID + j;
    bg[g] = bias[n];
#pragma unroll
    for (int d = 0; d < 6; ++d) wih6[g][d] = wih[n * 6 + d];
  }

  for (int i = tid; i < 32 * HID; i += 1024) ((_Float16*)lds_h)[i] = (_Float16)0.f;
  float cst[2][8];
#pragma unroll
  for (int mt = 0; mt < 2; ++mt)
#pragma unroll
    for (int v = 0; v < 8; ++v) cst[mt][v] = 0.f;
  __syncthreads();

  for (int t = 0; t < SEQ; ++t) {
    if (tid < 192) {
      int m = tid / 6, d = tid - m * 6;
      s_ein[m][d] = enc_in[((size_t)m * SEQ + t) * 6 + d];
    }
    __syncthreads();

    v8f acc[4][2];
#pragma unroll
    for (int g = 0; g < 4; ++g) { v8f z = {}; acc[g][0] = z; acc[g][1] = z; }

    for (int k = 0; k < HID; k += 32) {
      int kb = k + hi * 8;
      v16h a0 = load_a_h(&lds_h[laneLo][0], kb);
      v16h a1 = load_a_h(&lds_h[16 + laneLo][0], kb);
      int hk = k + hi * 16;
#pragma unroll
      for (int g = 0; g < 4; ++g) {
        v16h bm = load_b_h(whh + (size_t)(g * HID + j) * HID, hk);
        acc[g][0] = wmma16(a0, bm, acc[g][0]);
        acc[g][1] = wmma16(a1, bm, acc[g][1]);
      }
    }
    __syncthreads();  // all lds_h reads done before overwriting h

#pragma unroll
    for (int mt = 0; mt < 2; ++mt)
#pragma unroll
      for (int v = 0; v < 8; ++v) {
        int m = mt * 16 + v + hi * 8;
        float gi = acc[0][mt][v] + bg[0];
        float gf = acc[1][mt][v] + bg[1];
        float gg = acc[2][mt][v] + bg[2];
        float go = acc[3][mt][v] + bg[3];
#pragma unroll
        for (int d = 0; d < 6; ++d) {
          float xv = s_ein[m][d];
          gi += xv * wih6[0][d]; gf += xv * wih6[1][d];
          gg += xv * wih6[2][d]; go += xv * wih6[3][d];
        }
        float cn = sigm(gf) * cst[mt][v] + sigm(gi) * tanh_fast(gg);
        float hn = sigm(go) * tanh_fast(cn);
        cst[mt][v] = cn;
        lds_h[m][j] = (_Float16)hn;
        enc_out[((size_t)m * SEQ + t) * HID + j] = hn;
      }
    __syncthreads();
  }
}

// ctx[b,t,:] = sum_i attn[b,t,i] * enc_out[b,t-i,:]   (stored f16 for decoder)
__global__ __launch_bounds__(256)
void ctx_kernel(const float* __restrict__ attn, const float* __restrict__ enc_out,
                _Float16* __restrict__ ctx_h) {
  int bt = blockIdx.x;
  int b = bt >> 9, t = bt & 511;
  __shared__ float a[15];
  if (threadIdx.x < 15) a[threadIdx.x] = attn[(size_t)bt * 15 + threadIdx.x];
  __syncthreads();
  for (int j = threadIdx.x; j < HID; j += 256) {
    float s = 0.f;
#pragma unroll
    for (int i = 0; i < 15; ++i) {
      int tt = t - i;
      if (tt >= 0) s += a[i] * enc_out[((size_t)b * SEQ + tt) * HID + j];
    }
    ctx_h[(size_t)bt * HID + j] = (_Float16)s;
  }
}

// ---------------------------------------------------------------------------
// Decoder LSTM: persistent WG; fuses ctx @ Wih^T into the k-loop. ctx tile for
// step t+1 is async-prefetched into a double-buffered LDS tile during step t.
// ---------------------------------------------------------------------------
__global__ __launch_bounds__(1024)
void dec_lstm_kernel(const _Float16* __restrict__ ctx_h, const _Float16* __restrict__ whh,
                     const _Float16* __restrict__ wih, const float* __restrict__ bias,
                     const float* __restrict__ h0, const float* __restrict__ c0,
                     _Float16* __restrict__ dec_out) {
  __shared__ _Float16 lds_h[32][HID];
  __shared__ _Float16 s_ctx[2][32][HID];
  int tid = threadIdx.x;
  int lane = tid & 31, wv = tid >> 5;
  int laneLo = lane & 15, hi = lane >> 4;
  int j = wv * 16 + laneLo;

  // async-stage ctx[:, tt, :] (32x512 f16 = 32KB) into s_ctx[buf]: 32B/thread
  auto stage_ctx = [&](int tt, int buf) {
    int m = tid >> 5;
    int j0 = (tid & 31) * 16;
    const _Float16* src = ctx_h + ((size_t)m * SEQ + tt) * HID + j0;
    unsigned l = lds_off32(&s_ctx[buf][m][j0]);
    async_copy_b128(l, src);
    async_copy_b128(l + 16, src + 8);
  };

  float bg[4];
#pragma unroll
  for (int g = 0; g < 4; ++g) bg[g] = bias[g * HID + j];

  stage_ctx(0, 0);
  for (int i = tid; i < 32 * HID; i += 1024) {
    int jj = i & 511;
    ((_Float16*)lds_h)[i] = (_Float16)h0[jj];
  }
  float c0j = c0[j];
  float cst[2][8];
#pragma unroll
  for (int mt = 0; mt < 2; ++mt)
#pragma unroll
    for (int v = 0; v < 8; ++v) cst[mt][v] = c0j;
  wait_async0();
  __syncthreads();

  for (int t = 0; t < SEQ; ++t) {
    int buf = t & 1;
    if (t + 1 < SEQ) stage_ctx(t + 1, buf ^ 1);

    v8f acc[4][2];
#pragma unroll
    for (int g = 0; g < 4; ++g) { v8f z = {}; acc[g][0] = z; acc[g][1] = z; }

    const _Float16* x0 = &s_ctx[buf][laneLo][0];
    const _Float16* x1 = &s_ctx[buf][16 + laneLo][0];

    for (int k = 0; k < HID; k += 32) {
      int kb = k + hi * 8;
      v16h ah0 = load_a_h(&lds_h[laneLo][0], kb);
      v16h ah1 = load_a_h(&lds_h[16 + laneLo][0], kb);
      v16h ax0 = load_a_h(x0, kb);
      v16h ax1 = load_a_h(x1, kb);
      int hk = k + hi * 16;
#pragma unroll
      for (int g = 0; g < 4; ++g) {
        size_t nrow = (size_t)(g * HID + j) * HID;
        v16h bh = load_b_h(whh + nrow, hk);
        v16h bx = load_b_h(wih + nrow, hk);
        acc[g][0] = wmma16(ah0, bh, acc[g][0]);
        acc[g][1] = wmma16(ah1, bh, acc[g][1]);
        acc[g][0] = wmma16(ax0, bx, acc[g][0]);
        acc[g][1] = wmma16(ax1, bx, acc[g][1]);
      }
    }
    __syncthreads();

#pragma unroll
    for (int mt = 0; mt < 2; ++mt)
#pragma unroll
      for (int v = 0; v < 8; ++v) {
        int m = mt * 16 + v + hi * 8;
        float gi = acc[0][mt][v] + bg[0];
        float gf = acc[1][mt][v] + bg[1];
        float gg = acc[2][mt][v] + bg[2];
        float go = acc[3][mt][v] + bg[3];
        float cn = sigm(gf) * cst[mt][v] + sigm(gi) * tanh_fast(gg);
        float hn = sigm(go) * tanh_fast(cn);
        cst[mt][v] = cn;
        lds_h[m][j] = (_Float16)hn;
        dec_out[((size_t)m * SEQ + t) * HID + j] = (_Float16)hn;
      }
    wait_async0();   // next ctx tile has landed (per-wave), then publish
    __syncthreads();
  }
}

// hid = relu(dec_out @ W1^T + b1) -> f16
__global__ __launch_bounds__(256)
void mlp1_kernel(const _Float16* __restrict__ din, const _Float16* __restrict__ w1h,
                 const float* __restrict__ b1, _Float16* __restrict__ hid) {
  int tid = threadIdx.x;
  int lane = tid & 31, wv = tid >> 5;
  int laneLo = lane & 15, hi = lane >> 4;
  size_t row = (size_t)blockIdx.x * 16 + laneLo;
  const _Float16* arow = din + row * HID;

  v8f acc[4];
#pragma unroll
  for (int t = 0; t < 4; ++t) { v8f z = {}; acc[t] = z; }

  for (int k = 0; k < HID; k += 32) {
    int kb = k + hi * 8;
    v16h a = load_a_h(arow, kb);
    int hk = k + hi * 16;
    if (k + 64 < HID) __builtin_prefetch(w1h + (size_t)((wv * 4) * 16 + laneLo) * HID + k + 64, 0, 1);
#pragma unroll
    for (int t = 0; t < 4; ++t) {
      int n = (wv * 4 + t) * 16 + laneLo;
      acc[t] = wmma16(a, load_b_h(w1h + (size_t)n * HID, hk), acc[t]);
    }
  }
#pragma unroll
  for (int t = 0; t < 4; ++t) {
    int n = (wv * 4 + t) * 16 + laneLo;
    float bb = b1[n];
#pragma unroll
    for (int v = 0; v < 8; ++v) {
      int m = v + hi * 8;
      float x = acc[t][v] + bb;
      x = x > 0.f ? x : 0.f;
      hid[((size_t)blockIdx.x * 16 + m) * HID + n] = (_Float16)x;
    }
  }
}

// out[bt, o] = hid[bt,:] . W2[o,:] + b2[o]
__global__ void mlp2_kernel(const _Float16* __restrict__ hid, const float* __restrict__ W2,
                            const float* __restrict__ b2, float* __restrict__ out) {
  int idx = blockIdx.x * blockDim.x + threadIdx.x;
  if (idx >= MTOT * OUTD) return;
  int row = idx / OUTD, o = idx - row * OUTD;
  const _Float16* h = hid + (size_t)row * HID;
  const float* w = W2 + (size_t)o * HID;
  float s = b2[o];
  for (int n = 0; n < HID; n += 8) {
    v8h hv = *(const v8h*)(h + n);
#pragma unroll
    for (int i = 0; i < 8; ++i) s += (float)hv[i] * w[n + i];
  }
  out[idx] = s;
}

// ---------------------------------------------------------------------------
extern "C" void kernel_launch(void* const* d_in, const int* in_sizes, int n_in,
                              void* d_out, int out_size, void* d_ws, size_t ws_size,
                              hipStream_t stream) {
  (void)in_sizes; (void)n_in; (void)out_size; (void)ws_size;
  const float* face  = (const float*)d_in[1];
  const float* audio = (const float*)d_in[2];
  const float* descf = (const float*)d_in[3];
  const float* sitf  = (const float*)d_in[4];
  const float* scene = (const float*)d_in[5];
  const float* text  = (const float*)d_in[6];
  const float* Wt  = (const float*)d_in[8];  const float* btv  = (const float*)d_in[9];
  const float* Wf  = (const float*)d_in[10]; const float* bfv  = (const float*)d_in[11];
  const float* Wa  = (const float*)d_in[12]; const float* bav  = (const float*)d_in[13];
  const float* Wsc = (const float*)d_in[14]; const float* bscv = (const float*)d_in[15];
  const float* Wd  = (const float*)d_in[16]; const float* bdv  = (const float*)d_in[17];
  const float* Wsi = (const float*)d_in[18]; const float* bsiv = (const float*)d_in[19];
  const float* w_att = (const float*)d_in[20]; const float* b_att = (const float*)d_in[21];
  const float* w_ut  = (const float*)d_in[22]; const float* b_ut  = (const float*)d_in[23];
  const float* w_uf  = (const float*)d_in[24]; const float* b_uf  = (const float*)d_in[25];
  const float* w_ua  = (const float*)d_in[26]; const float* b_ua  = (const float*)d_in[27];
  const float* eWih = (const float*)d_in[28]; const float* eWhh = (const float*)d_in[29];
  const float* eb   = (const float*)d_in[30];
  const float* dWih = (const float*)d_in[31]; const float* dWhh = (const float*)d_in[32];
  const float* db   = (const float*)d_in[33];
  const float* dh0  = (const float*)d_in[34]; const float* dc0 = (const float*)d_in[35];
  const float* W1   = (const float*)d_in[36]; const float* b1  = (const float*)d_in[37];
  const float* W2   = (const float*)d_in[38]; const float* b2  = (const float*)d_in[39];
  const float* w1v = w_att;
  const float* w2v = w_att + HID;

  char* ws = (char*)d_ws;
  size_t off = 0;
  auto alloc = [&](size_t bytes) -> char* {
    char* p = ws + off;
    off = (off + bytes + 255) & ~(size_t)255;
    return p;
  };

  _Float16* wt_h  = (_Float16*)alloc((size_t)512 * 768  * 2);
  _Float16* wf_h  = (_Float16*)alloc((size_t)512 * 512  * 2);
  _Float16* wa_h  = (_Float16*)alloc((size_t)512 * 128  * 2);
  _Float16* wsc_h = (_Float16*)alloc((size_t)512 * 2048 * 2);
  _Float16* wd_h  = (_Float16*)alloc((size_t)512 * 768  * 2);
  _Float16* wsi_h = (_Float16*)alloc((size_t)512 * 768  * 2);
  _Float16* ewhh_h= (_Float16*)alloc((size_t)2048 * 512 * 2);
  _Float16* dwih_h= (_Float16*)alloc((size_t)2048 * 512 * 2);
  _Float16* dwhh_h= (_Float16*)alloc((size_t)2048 * 512 * 2);
  _Float16* w1_h  = (_Float16*)alloc((size_t)512 * 512  * 2);
  float* S        = (float*)alloc((size_t)18 * MTOT * 4);
  float* att_raw  = (float*)alloc((size_t)15 * MTOT * 4);
  float* uni_raw  = (float*)alloc((size_t)6  * MTOT * 4);
  float* attn     = (float*)alloc((size_t)15 * MTOT * 4);
  float* enc_in   = (float*)alloc((size_t)6  * MTOT * 4);
  float* enc_out  = (float*)alloc((size_t)MTOT * HID * 4);
  _Float16* ctx_h = (_Float16*)alloc((size_t)MTOT * HID * 2);
  _Float16* dec_o = (_Float16*)alloc((size_t)MTOT * HID * 2);
  _Float16* hid_h = (_Float16*)alloc((size_t)MTOT * HID * 2);

  auto cvt = [&](const float* s, _Float16* d, int n) {
    cvt_f32_f16<<<(n + 255) / 256, 256, 0, stream>>>(s, d, n);
  };
  cvt(Wt,   wt_h,   512 * 768);
  cvt(Wf,   wf_h,   512 * 512);
  cvt(Wa,   wa_h,   512 * 128);
  cvt(Wsc,  wsc_h,  512 * 2048);
  cvt(Wd,   wd_h,   512 * 768);
  cvt(Wsi,  wsi_h,  512 * 768);
  cvt(eWhh, ewhh_h, 2048 * 512);
  cvt(dWih, dwih_h, 2048 * 512);
  cvt(dWhh, dwhh_h, 2048 * 512);
  cvt(W1,   w1_h,   512 * 512);

  const int nblk = MTOT / 16;
  proj_kernel<<<nblk, 256, 0, stream>>>(text,  wt_h,  btv,  w1v, w2v, w_ut,  768,
                                        S + 0*MTOT, S + 1*MTOT, S + 2*MTOT);
  proj_kernel<<<nblk, 256, 0, stream>>>(face,  wf_h,  bfv,  w1v, w2v, w_uf,  512,
                                        S + 3*MTOT, S + 4*MTOT, S + 5*MTOT);
  proj_kernel<<<nblk, 256, 0, stream>>>(audio, wa_h,  bav,  w1v, w2v, w_ua,  128,
                                        S + 6*MTOT, S + 7*MTOT, S + 8*MTOT);
  proj_kernel<<<nblk, 256, 0, stream>>>(scene, wsc_h, bscv, w1v, w2v, w_ut, 2048,
                                        S + 9*MTOT, S + 10*MTOT, S + 11*MTOT);
  proj_kernel<<<nblk, 256, 0, stream>>>(descf, wd_h,  bdv,  w1v, w2v, w_uf,  768,
                                        S + 12*MTOT, S + 13*MTOT, S + 14*MTOT);
  proj_kernel<<<nblk, 256, 0, stream>>>(sitf,  wsi_h, bsiv, w1v, w2v, w_ua,  768,
                                        S + 15*MTOT, S + 16*MTOT, S + 17*MTOT);

  softmax_kernel<<<dim3(21, 32), 256, 0, stream>>>(S, b_att, b_ut, b_uf, b_ua,
                                                   att_raw, uni_raw);
  gather_kernel<<<(MTOT + 255) / 256, 256, 0, stream>>>(att_raw, uni_raw, attn, enc_in);

  enc_lstm_kernel<<<1, 1024, 0, stream>>>(enc_in, ewhh_h, eWih, eb, enc_out);
  ctx_kernel<<<MTOT, 256, 0, stream>>>(attn, enc_out, ctx_h);
  dec_lstm_kernel<<<1, 1024, 0, stream>>>(ctx_h, dwhh_h, dwih_h, db, dh0, dc0, dec_o);

  mlp1_kernel<<<nblk, 256, 0, stream>>>(dec_o, w1_h, b1, hid_h);
  mlp2_kernel<<<(MTOT * OUTD + 255) / 256, 256, 0, stream>>>(hid_h, W2, b2, (float*)d_out);
}
